// SparseFrameAttentionModule_72275709657149
// MI455X (gfx1250) — matrix operand order repaired and checked by reading, not compiled, not run on hardware
//
#include <hip/hip_runtime.h>
#include <hip/hip_bf16.h>

#define NUM_HEADS   12
#define NUM_FRAMES  42
#define FRAME_HW    390
#define TOP_K       5
#define CHUNK_SIZE  5
#define TARGET_FR   21
#define TOTAL_SEL   10
#define DIMC        768
#define BATCHN      2
#define HEAD_DIM    64
#define SEQ         (NUM_FRAMES*FRAME_HW)      // 16380
#define KEYS        (TOTAL_SEL*FRAME_HW)       // 3900
#define KTILE       64
#define NFULL       (KEYS / KTILE)             // 60 full rounds (3840 keys)
#define QB          4                           // 4 blocks x 8 waves = 32 >= 25 q-tiles

typedef __attribute__((ext_vector_type(16))) _Float16 v16h;
typedef __attribute__((ext_vector_type(8)))  _Float16 v8h;
typedef __attribute__((ext_vector_type(4)))  _Float16 v4h;
typedef __attribute__((ext_vector_type(8)))  float    v8f;

__device__ __forceinline__ v8f wmma_f16(v16h a, v16h b, v8f c) {
  // D(16x16,f32) = A(16x32,f16) * B(32x16,f16) + C
  return __builtin_amdgcn_wmma_f32_16x16x32_f16(false, a, false, b, (short)0, c, false, false);
}

// ---------------- Phase 1: per-frame mean representations -------------------
__global__ void repr_kernel(const float* __restrict__ q, const float* __restrict__ k,
                            float* __restrict__ qrepr, float* __restrict__ krepr) {
  const int bf = blockIdx.x;                 // b*NUM_FRAMES + f
  const float inv = 1.0f / (float)FRAME_HW;
  for (int d = threadIdx.x; d < DIMC; d += blockDim.x) {
    const float* qp = q + (size_t)bf * FRAME_HW * DIMC + d;
    const float* kp = k + (size_t)bf * FRAME_HW * DIMC + d;
    float sq = 0.f, sk = 0.f;
    for (int r = 0; r < FRAME_HW; ++r) { sq += qp[(size_t)r * DIMC]; sk += kp[(size_t)r * DIMC]; }
    qrepr[bf * DIMC + d] = sq * inv;
    krepr[bf * DIMC + d] = sk * inv;
  }
}

// ---------------- Phase 2: sim(21x21) + top-k + sel build -------------------
__global__ void sel_kernel(const float* __restrict__ qrepr, const float* __restrict__ krepr,
                           const int* __restrict__ cam, int* __restrict__ sel) {
  __shared__ float sim[TARGET_FR][TARGET_FR];
  __shared__ int   topk[TARGET_FR][TOP_K];
  const int b = blockIdx.x;
  const int tid = threadIdx.x;               // 128 threads
  for (int p = tid; p < TARGET_FR * TARGET_FR; p += blockDim.x) {
    const int fr = p / TARGET_FR, g = p % TARGET_FR;
    const float* qp = qrepr + (size_t)(b * NUM_FRAMES + TARGET_FR + fr) * DIMC;
    const float* kp = krepr + (size_t)(b * NUM_FRAMES + TARGET_FR + g) * DIMC;
    float s = 0.f;
    for (int d = 0; d < DIMC; ++d) s += qp[d] * kp[d];
    sim[fr][g] = s;
  }
  __syncthreads();
  if (tid < TARGET_FR) {
    float vals[TARGET_FR]; bool used[TARGET_FR];
    for (int g = 0; g < TARGET_FR; ++g) { vals[g] = sim[tid][g]; used[g] = false; }
    for (int j = 0; j < TOP_K; ++j) {
      int best = 0; float bv = -__builtin_inff();
      for (int g = 0; g < TARGET_FR; ++g)
        if (!used[g] && vals[g] > bv) { bv = vals[g]; best = g; }
      used[best] = true;
      topk[tid][j] = best + TARGET_FR;
    }
  }
  __syncthreads();
  if (tid < NUM_FRAMES) {
    const int f = tid;
    int start;
    if (f <= 18) { start = f - 2; if (start < 0) start = 0; } else { start = 16; }
    int* sp = sel + (b * NUM_FRAMES + f) * TOTAL_SEL;
    for (int j = 0; j < CHUNK_SIZE; ++j) sp[j] = start + j;
    for (int j = 0; j < TOP_K; ++j)
      sp[CHUNK_SIZE + j] = (f < TARGET_FR) ? cam[f * TOP_K + j] : topk[f - TARGET_FR][j];
  }
}

// ---------------- Phase 3: flash attention with WMMA ------------------------
// One 64-key round: stage K/V -> 8 S-WMMAs -> online softmax -> 8 O-WMMAs.
template<bool TAIL>
__device__ __forceinline__ void attn_round(
    const float* __restrict__ kk, const float* __restrict__ vv,
    const int* selF, int kbase, int b, int h,
    int tid, int lm, int half, int hoff,
    _Float16* ldsK, _Float16* ldsVT, _Float16* pp,
    const v16h& aQ0, const v16h& aQ1,
    float* mrow, float* lrow, v8f* oacc)
{
  __syncthreads();  // previous round's consumers done

  // ---- stage K (row major, packed b64) and V (transposed) as f16 ----
  #pragma unroll
  for (int e = 0; e < 4; ++e) {
    const int pidx = e * 256 + tid;          // float4-quad index, 0..1023
    const int key  = pidx >> 4;              // 0..63
    const int dq   = (pidx & 15) << 2;       // 0,4,...,60
    const int gk   = kbase + key;
    float4 kq = {0.f,0.f,0.f,0.f}, vq = {0.f,0.f,0.f,0.f};
    if (!TAIL || gk < KEYS) {
      const int slot = gk / FRAME_HW;
      const int row  = gk - slot * FRAME_HW;
      const size_t base = ((size_t)b * SEQ + (size_t)selF[slot] * FRAME_HW + row) * DIMC
                          + h * HEAD_DIM + dq;
      kq = *(const float4*)(kk + base);
      vq = *(const float4*)(vv + base);
    }
    const v4h kp4 = { (_Float16)kq.x, (_Float16)kq.y, (_Float16)kq.z, (_Float16)kq.w };
    *(v4h*)&ldsK[key * HEAD_DIM + dq] = kp4;
    ldsVT[(dq + 0) * KTILE + key] = (_Float16)vq.x;
    ldsVT[(dq + 1) * KTILE + key] = (_Float16)vq.y;
    ldsVT[(dq + 2) * KTILE + key] = (_Float16)vq.z;
    ldsVT[(dq + 3) * KTILE + key] = (_Float16)vq.w;
  }
  __syncthreads();

  // ---- S = Q * K^T : four 16-key tiles, d split into 2x32 ----
  v8f x[4];
  #pragma unroll
  for (int st = 0; st < 4; ++st) {
    const _Float16* kp = &ldsK[(st * 16 + lm) * HEAD_DIM];
    const v16h bK0 = *(const v16h*)(kp + half * 16);
    const v16h bK1 = *(const v16h*)(kp + 32 + half * 16);
    v8f s = {0.f,0.f,0.f,0.f,0.f,0.f,0.f,0.f};
    s = wmma_f16(aQ0, bK0, s);
    s = wmma_f16(aQ1, bK1, s);
    if (TAIL) {
      const bool valid = (kbase + st * 16 + lm) < KEYS;
      #pragma unroll
      for (int r = 0; r < 8; ++r) x[st][r] = valid ? s[r] : -__builtin_inff();
    } else {
      x[st] = s;
    }
  }

  // ---- online softmax: row = vgpr r + 8*half, cols across 16 lanes ----
  v8f t;
  #pragma unroll
  for (int r = 0; r < 8; ++r)
    t[r] = fmaxf(fmaxf(x[0][r], x[1][r]), fmaxf(x[2][r], x[3][r]));
  #pragma unroll
  for (int msk = 1; msk < 16; msk <<= 1)
    #pragma unroll
    for (int r = 0; r < 8; ++r) t[r] = fmaxf(t[r], __shfl_xor(t[r], msk, 32));

  float alpha[8];
  #pragma unroll
  for (int r = 0; r < 8; ++r) {
    const float mn = fmaxf(mrow[r], t[r]);
    alpha[r] = exp2f(mrow[r] - mn);
    mrow[r] = mn;
  }
  v8f p[4], ssum;
  #pragma unroll
  for (int st = 0; st < 4; ++st)
    #pragma unroll
    for (int r = 0; r < 8; ++r) p[st][r] = exp2f(x[st][r] - mrow[r]);
  #pragma unroll
  for (int r = 0; r < 8; ++r)
    ssum[r] = (p[0][r] + p[1][r]) + (p[2][r] + p[3][r]);
  #pragma unroll
  for (int msk = 1; msk < 16; msk <<= 1)
    #pragma unroll
    for (int r = 0; r < 8; ++r) ssum[r] += __shfl_xor(ssum[r], msk, 32);
  #pragma unroll
  for (int r = 0; r < 8; ++r) lrow[r] = lrow[r] * alpha[r] + ssum[r];
  #pragma unroll
  for (int dc = 0; dc < 4; ++dc)
    #pragma unroll
    for (int r = 0; r < 8; ++r) oacc[dc][r] *= alpha[r];

  // ---- bounce P (C/D layout) -> A-matrix layout via per-wave LDS slab ----
  #pragma unroll
  for (int st = 0; st < 4; ++st)
    #pragma unroll
    for (int r = 0; r < 8; ++r)
      pp[(r + 8 * half) * KTILE + st * 16 + lm] = (_Float16)p[st][r];
  asm volatile("s_wait_dscnt 0" ::: "memory");  // per-wave region: wait, no barrier

  const _Float16* pr = pp + lm * KTILE;
  const v16h aP0 = __builtin_shufflevector(*(const v8h*)(pr + 2 * hoff),
                                           *(const v8h*)(pr + 8 + 2 * hoff),
                                           0,1,2,3,4,5,6,7,8,9,10,11,12,13,14,15);
  const v16h aP1 = __builtin_shufflevector(*(const v8h*)(pr + 32 + 2 * hoff),
                                           *(const v8h*)(pr + 40 + 2 * hoff),
                                           0,1,2,3,4,5,6,7,8,9,10,11,12,13,14,15);

  // ---- O += P * V (4 d-chunks of 16, 2 key-chunks of 32) ----
  #pragma unroll
  for (int dc = 0; dc < 4; ++dc) {
    const _Float16* vr = &ldsVT[(dc * 16 + lm) * KTILE];
    const v16h bV0 = *(const v16h*)(vr + half * 16);
    const v16h bV1 = *(const v16h*)(vr + 32 + half * 16);
    oacc[dc] = wmma_f16(aP0, bV0, oacc[dc]);
    oacc[dc] = wmma_f16(aP1, bV1, oacc[dc]);
  }
}

// block = 256 threads = 8 waves; each wave owns one 16-row Q tile.
// grid = BATCH * NUM_FRAMES * NUM_HEADS * QB
__global__ __launch_bounds__(256) void attn_kernel(const float* __restrict__ q,
                                                   const float* __restrict__ k,
                                                   const float* __restrict__ v,
                                                   const int* __restrict__ sel,
                                                   float* __restrict__ out) {
  __shared__ __align__(32) _Float16 ldsK[KTILE * HEAD_DIM];   // [key][d]     8KB
  __shared__ __align__(32) _Float16 ldsVT[HEAD_DIM * KTILE];  // [d][key]     8KB
  __shared__ __align__(32) _Float16 ldsP[8 * 16 * KTILE];     // per-wave P  16KB
  __shared__ int selF[TOTAL_SEL];

  int tmp = blockIdx.x;
  const int qb = tmp & (QB - 1); tmp >>= 2;
  const int h  = tmp % NUM_HEADS; tmp /= NUM_HEADS;
  const int f  = tmp % NUM_FRAMES;
  const int b  = tmp / NUM_FRAMES;

  const int tid  = threadIdx.x;
  const int w    = tid >> 5;
  const int lane = tid & 31;
  const int lm   = lane & 15;
  const int half = lane >> 4;
  const int hoff = half * 8;

  if (tid < TOTAL_SEL) selF[tid] = sel[(b * NUM_FRAMES + f) * TOTAL_SEL + tid];

  const int qtile = qb * 8 + w;
  const int qrow0 = qtile * 16;

  // ---- Load Q into A-matrix layout, pre-scaled by hd^-0.5 * log2(e) ----
  const float qsc = 0.125f * 1.44269504088896340736f;
  v16h aQ0, aQ1;
  {
    const int qr = qrow0 + lm;
    if (qr < FRAME_HW) {
      const float* qp = q + ((size_t)b * SEQ + (size_t)f * FRAME_HW + qr) * DIMC + h * HEAD_DIM;
      #pragma unroll
      for (int e = 0; e < 8; ++e) {
        aQ0[e]     = (_Float16)(qp[2 * hoff + e] * qsc);
        aQ0[8 + e] = (_Float16)(qp[8 + 2 * hoff + e] * qsc);
        aQ1[e]     = (_Float16)(qp[32 + 2 * hoff + e] * qsc);
        aQ1[8 + e] = (_Float16)(qp[40 + 2 * hoff + e] * qsc);
      }
    } else {
      #pragma unroll
      for (int e = 0; e < 16; ++e) { aQ0[e] = (_Float16)0.f; aQ1[e] = (_Float16)0.f; }
    }
  }

  float mrow[8], lrow[8];
  v8f oacc[4];
  #pragma unroll
  for (int r = 0; r < 8; ++r) { mrow[r] = -__builtin_inff(); lrow[r] = 0.f; }
  #pragma unroll
  for (int dc = 0; dc < 4; ++dc)
    #pragma unroll
    for (int r = 0; r < 8; ++r) oacc[dc][r] = 0.f;

  _Float16* pp = &ldsP[w * 16 * KTILE];

  for (int kt = 0; kt < NFULL; ++kt)
    attn_round<false>(k, v, selF, kt * KTILE, b, h, tid, lm, half, hoff,
                      ldsK, ldsVT, pp, aQ0, aQ1, mrow, lrow, oacc);
  attn_round<true>(k, v, selF, NFULL * KTILE, b, h, tid, lm, half, hoff,
                   ldsK, ldsVT, pp, aQ0, aQ1, mrow, lrow, oacc);

  // ---- normalize and store ----
  #pragma unroll
  for (int r = 0; r < 8; ++r) {
    const int qr = qrow0 + r + 8 * half;
    if (qr < FRAME_HW) {
      const float inv = 1.0f / lrow[r];
      float* op = out + ((size_t)b * SEQ + (size_t)f * FRAME_HW + qr) * DIMC
                  + h * HEAD_DIM + lm;
      op[0]  = oacc[0][r] * inv;
      op[16] = oacc[1][r] * inv;
      op[32] = oacc[2][r] * inv;
      op[48] = oacc[3][r] * inv;
    }
  }
}

extern "C" void kernel_launch(void* const* d_in, const int* in_sizes, int n_in,
                              void* d_out, int out_size, void* d_ws, size_t ws_size,
                              hipStream_t stream) {
  const float* q   = (const float*)d_in[0];
  const float* k   = (const float*)d_in[1];
  const float* v   = (const float*)d_in[2];
  const int*   cam = (const int*)d_in[3];
  float* out = (float*)d_out;

  float* qrepr = (float*)d_ws;                               // B*F*DIM
  float* krepr = qrepr + BATCHN * NUM_FRAMES * DIMC;         // B*F*DIM
  int*   sel   = (int*)(krepr + BATCHN * NUM_FRAMES * DIMC); // B*F*10

  repr_kernel<<<BATCHN * NUM_FRAMES, 256, 0, stream>>>(q, k, qrepr, krepr);
  sel_kernel<<<BATCHN, 128, 0, stream>>>(qrepr, krepr, cam, sel);
  attn_kernel<<<BATCHN * NUM_FRAMES * NUM_HEADS * QB, 256, 0, stream>>>(q, k, v, sel, out);
}
